// CrystalNet_42949673432
// MI455X (gfx1250) — compile-verified, matrix operand order checked by reading, not run on hardware
//
#include <hip/hip_runtime.h>

// ---------------- types ----------------
typedef __bf16 bf16_t;
typedef __attribute__((ext_vector_type(16))) __bf16 v16bf;
typedef __attribute__((ext_vector_type(8)))  __bf16 v8bf;
typedef __attribute__((ext_vector_type(8)))  float  v8f;

#define NB   2
#define SEQ  1024
#define DIM  1024
#define NH   8
#define DH   128
#define NIT  28
#define VOC  50257
#define MTOT (NB*SEQ)           // 2048
#define EPSV 1.1920929e-07f
#define NEGBIG -3.0e38f

static __device__ __forceinline__ bf16_t f2bf(float f) {
    unsigned u = __builtin_bit_cast(unsigned, f);
    unsigned r = u + 0x7FFFu + ((u >> 16) & 1u);
    unsigned short h = (unsigned short)(r >> 16);
    return __builtin_bit_cast(bf16_t, h);
}

// Load a 16-bit A/B WMMA fragment half-pair: elements [0..7]=src[0..7], [8..15]=src[16..23].
// Caller passes p = &Mtx[row*ld + kBase + (lane>>4)*8].
static __device__ __forceinline__ v16bf ld_frag(const bf16_t* p) {
    const v8bf* q = (const v8bf*)p;
    v8bf lo = q[0];
    v8bf hi = q[2];   // +16 elements
    return __builtin_shufflevector(lo, hi, 0,1,2,3,4,5,6,7,8,9,10,11,12,13,14,15);
}

static __device__ __forceinline__ v8f wmma_bf16(v16bf a, v16bf b, v8f c) {
    return __builtin_amdgcn_wmma_f32_16x16x32_bf16(false, a, false, b, (short)0, c, false, false);
}

// ---------------- elementwise kernels ----------------
__global__ __launch_bounds__(256) void k_embed(const int* tok, const float* emb, float* x) {
    int row = blockIdx.x;                    // 0..MTOT-1
    int t = tok[row];
    const float4* src = (const float4*)(emb + (size_t)t * DIM);
    float4* dst = (float4*)(x + (size_t)row * DIM);
    dst[threadIdx.x] = src[threadIdx.x];     // 256 * 4 = 1024
}

__global__ __launch_bounds__(256) void k_cvt_bf16(const float* src, bf16_t* dst, int n) {
    for (int i = blockIdx.x * 256 + threadIdx.x; i < n; i += gridDim.x * 256)
        dst[i] = f2bf(src[i]);
}

__global__ __launch_bounds__(256) void k_add_bf16(const float* a, const float* b, bf16_t* dst, int n) {
    int i = blockIdx.x * 256 + threadIdx.x;
    if (i < n) dst[i] = f2bf(a[i] + b[i]);
}

// Vb[b,s,h*DH+d] -> Vt[((b*NH+h)*DH+d)*SEQ + s]
__global__ __launch_bounds__(256) void k_transposeV(const bf16_t* Vb, bf16_t* Vt) {
    int idx = blockIdx.x * 256 + threadIdx.x;        // over NB*NH*DH*SEQ = 2M
    int s = idx & (SEQ - 1);
    int d = (idx >> 10) & (DH - 1);
    int h = (idx >> 17) & (NH - 1);
    int b = idx >> 20;
    Vt[idx] = Vb[((size_t)(b * SEQ + s)) * DIM + h * DH + d];
}

// h = rmsnorm(x, nw) [* g + bt]  -> bf16
__global__ __launch_bounds__(256) void k_rmsnorm(const float* x, const float* nw,
                                                 const float* g, const float* bt, bf16_t* out) {
    int row = blockIdx.x;
    int c = threadIdx.x * 4;
    const float* xr = x + (size_t)row * DIM;
    float4 xv = *(const float4*)(xr + c);
    float ss = xv.x * xv.x + xv.y * xv.y + xv.z * xv.z + xv.w * xv.w;
    for (int o = 16; o; o >>= 1) ss += __shfl_xor(ss, o, 32);
    __shared__ float sp[8];
    int lane = threadIdx.x & 31, wave = threadIdx.x >> 5;
    if (lane == 0) sp[wave] = ss;
    __syncthreads();
    float tot = 0.f;
    for (int w = 0; w < 8; w++) tot += sp[w];
    float inv = rsqrtf(tot * (1.0f / DIM) + EPSV);
    bf16_t* orow = out + (size_t)row * DIM;
    float hv[4] = { xv.x, xv.y, xv.z, xv.w };
    for (int i = 0; i < 4; i++) {
        float v = hv[i] * inv * nw[c + i];
        if (g) v = v * g[c + i] + bt[c + i];
        orow[c + i] = f2bf(v);
    }
}

// ---------------- GEMM: C = A[M,K](bf16) x B[N,K](bf16)^T ----------------
// 128 threads = 4 waves (2M x 2N), wave tile 32x64, WG tile 64x128.
// Per k-step: 6 fragment loads (12 b128) feed 8 WMMAs -> 1.5 loads/WMMA.
__global__ __launch_bounds__(128) void k_gemm(const bf16_t* __restrict__ A,
                                              const bf16_t* __restrict__ Bw,
                                              int N, int K,
                                              bf16_t* out_bf, float* out_f32,
                                              float* accum, const float* step) {
    int lane = threadIdx.x & 31;
    int wave = threadIdx.x >> 5;
    int wm = wave >> 1, wn = wave & 1;
    int tm = blockIdx.y * 64 + wm * 32;
    int tn = blockIdx.x * 128 + wn * 64;
    int l15 = lane & 15, lh = lane >> 4;

    v8f c[2][4] = {};

    int ar0 = tm + l15, ar1 = tm + 16 + l15;
    const bf16_t* Ap0 = A + (size_t)ar0 * K;
    const bf16_t* Ap1 = A + (size_t)ar1 * K;
    const bf16_t* Bp[4];
    for (int j = 0; j < 4; j++) {
        int bn = tn + j * 16 + l15;
        Bp[j] = Bw + (size_t)(bn < N ? bn : N - 1) * K;
    }

    for (int k0 = 0; k0 < K; k0 += 32) {
        int kb = k0 + lh * 8;
        v16bf a0 = ld_frag(Ap0 + kb);
        v16bf a1 = ld_frag(Ap1 + kb);
        v16bf b0 = ld_frag(Bp[0] + kb);
        v16bf b1 = ld_frag(Bp[1] + kb);
        v16bf b2 = ld_frag(Bp[2] + kb);
        v16bf b3 = ld_frag(Bp[3] + kb);
        c[0][0] = wmma_bf16(a0, b0, c[0][0]);
        c[0][1] = wmma_bf16(a0, b1, c[0][1]);
        c[0][2] = wmma_bf16(a0, b2, c[0][2]);
        c[0][3] = wmma_bf16(a0, b3, c[0][3]);
        c[1][0] = wmma_bf16(a1, b0, c[1][0]);
        c[1][1] = wmma_bf16(a1, b1, c[1][1]);
        c[1][2] = wmma_bf16(a1, b2, c[1][2]);
        c[1][3] = wmma_bf16(a1, b3, c[1][3]);
    }

    float s = (accum && step) ? *step : 1.0f;
    for (int mi = 0; mi < 2; mi++) {
        for (int ni = 0; ni < 4; ni++) {
            int col = tn + ni * 16 + l15;
            if (col >= N) continue;
            v8f cc = c[mi][ni];
            int rb = tm + mi * 16 + lh * 8;
            for (int r = 0; r < 8; r++) {
                size_t idx = (size_t)(rb + r) * N + col;
                float v = cc[r];
                if (out_bf)  out_bf[idx] = f2bf(v);
                if (out_f32) out_f32[idx] = v;
                if (accum)   accum[idx] += v * s;
            }
        }
    }
}

// ---------------- attention: softmax(-scale*K K^T, causal) V ----------------
// grid (SEQ/16, NH, NB), one wave per block.
__global__ __launch_bounds__(32) void k_attn(const bf16_t* __restrict__ Kb,
                                             const bf16_t* __restrict__ Vt,
                                             float* __restrict__ Pout) {
    const float scale = 0.088388347648318447f;   // 1/sqrt(128)
    int it = blockIdx.x, h = blockIdx.y, b = blockIdx.z;
    int lane = threadIdx.x;
    int l15 = lane & 15, lh = lane >> 4;

    __shared__ bf16_t Plds[16 * 32];

    // persistent A fragments: K rows of this i-tile
    int irow = it * 16 + l15;
    const bf16_t* Ki = Kb + ((size_t)(b * SEQ + irow)) * DIM + h * DH;
    v16bf aK[4];
    for (int ks = 0; ks < 4; ks++) aK[ks] = ld_frag(Ki + ks * 32 + lh * 8);

    float m[8], l[8];
    v8f O[8];
    for (int r = 0; r < 8; r++) { m[r] = NEGBIG; l[r] = 0.f; }
    for (int cch = 0; cch < 8; cch++)
        for (int r = 0; r < 8; r++) O[cch][r] = 0.f;

    int imax = it * 16 + 15;
    for (int j0 = 0; j0 <= imax; j0 += 32) {
        // ---- energy tiles (16 i x 32 j) ----
        v8f e0 = {}, e1 = {};
        const bf16_t* Kj0 = Kb + ((size_t)(b * SEQ + j0 + l15)) * DIM + h * DH;
        const bf16_t* Kj1 = Kb + ((size_t)(b * SEQ + j0 + 16 + l15)) * DIM + h * DH;
        for (int ks = 0; ks < 4; ks++) {
            e0 = wmma_bf16(aK[ks], ld_frag(Kj0 + ks * 32 + lh * 8), e0);
            e1 = wmma_bf16(aK[ks], ld_frag(Kj1 + ks * 32 + lh * 8), e1);
        }

        // ---- online softmax ----
        float av[8];
        for (int r = 0; r < 8; r++) {
            int i = it * 16 + lh * 8 + r;
            float lg0 = (j0 + l15      <= i) ? (-scale * e0[r]) : NEGBIG;
            float lg1 = (j0 + 16 + l15 <= i) ? (-scale * e1[r]) : NEGBIG;
            float mx = fmaxf(lg0, lg1);
            for (int o = 8; o; o >>= 1) mx = fmaxf(mx, __shfl_xor(mx, o, 16));
            float mn = fmaxf(m[r], mx);
            float al = __expf(m[r] - mn);
            float p0 = (lg0 > NEGBIG) ? __expf(lg0 - mn) : 0.f;
            float p1 = (lg1 > NEGBIG) ? __expf(lg1 - mn) : 0.f;
            float rs = p0 + p1;
            for (int o = 8; o; o >>= 1) rs += __shfl_xor(rs, o, 16);
            l[r] = l[r] * al + rs;
            m[r] = mn;
            av[r] = al;
            Plds[(lh * 8 + r) * 32 + l15]      = f2bf(p0);
            Plds[(lh * 8 + r) * 32 + 16 + l15] = f2bf(p1);
        }
        for (int cch = 0; cch < 8; cch++)
            for (int r = 0; r < 8; r++) O[cch][r] *= av[r];

        __syncthreads();

        // ---- P x V ----
        v16bf aP = ld_frag(Plds + l15 * 32 + lh * 8);
        const bf16_t* Vb0 = Vt + (((size_t)(b * NH + h)) * DH) * SEQ + j0 + lh * 8;
        for (int cch = 0; cch < 8; cch++) {
            int d = cch * 16 + l15;
            O[cch] = wmma_bf16(aP, ld_frag(Vb0 + (size_t)d * SEQ), O[cch]);
        }
        __syncthreads();
    }

    // ---- normalize + store fp32 ----
    for (int cch = 0; cch < 8; cch++) {
        int col = h * DH + cch * 16 + l15;
        for (int r = 0; r < 8; r++) {
            int row = it * 16 + lh * 8 + r;
            Pout[((size_t)(b * SEQ + row)) * DIM + col] = O[cch][r] / l[r];
        }
    }
}

// ---------------- host launcher ----------------
extern "C" void kernel_launch(void* const* d_in, const int* in_sizes, int n_in,
                              void* d_out, int out_size, void* d_ws, size_t ws_size,
                              hipStream_t stream) {
    const int*   tokens    = (const int*)  d_in[0];
    const float* embed     = (const float*)d_in[1];
    const float* Wk        = (const float*)d_in[2];
    const float* Wv        = (const float*)d_in[3];
    const float* Ws        = (const float*)d_in[4];
    const float* Wo        = (const float*)d_in[5];
    const float* step_size = (const float*)d_in[6];
    const float* norm_w    = (const float*)d_in[7];
    const float* gamma     = (const float*)d_in[8];
    const float* beta      = (const float*)d_in[9];
    const float* out_norm  = (const float*)d_in[10];
    const float* lm_head   = (const float*)d_in[11];
    float* out = (float*)d_out;

    // workspace carve
    char* w = (char*)d_ws;
    auto alloc = [&](size_t bytes) -> void* {
        void* p = (void*)w;
        w += (bytes + 255) & ~(size_t)255;
        return p;
    };
    float*  x    = (float*) alloc((size_t)MTOT * DIM * 4);
    bf16_t* hb   = (bf16_t*)alloc((size_t)MTOT * DIM * 2);
    bf16_t* Kb   = (bf16_t*)alloc((size_t)MTOT * DIM * 2);
    bf16_t* Vb   = (bf16_t*)alloc((size_t)MTOT * DIM * 2);
    bf16_t* Vt   = (bf16_t*)alloc((size_t)MTOT * DIM * 2);
    float*  SWb  = (float*) alloc((size_t)MTOT * DIM * 4);
    float*  Pb   = (float*) alloc((size_t)MTOT * DIM * 4);
    bf16_t* Tb   = (bf16_t*)alloc((size_t)MTOT * DIM * 2);
    bf16_t* xb   = (bf16_t*)alloc((size_t)MTOT * DIM * 2);
    bf16_t* Wk16 = (bf16_t*)alloc((size_t)DIM * DIM * 2);
    bf16_t* Wv16 = (bf16_t*)alloc((size_t)DIM * DIM * 2);
    bf16_t* Ws16 = (bf16_t*)alloc((size_t)DIM * DIM * 2);
    bf16_t* Wo16 = (bf16_t*)alloc((size_t)DIM * DIM * 2);
    bf16_t* LM16 = (bf16_t*)alloc((size_t)VOC * DIM * 2);

    const int nDD = DIM * DIM;
    const int nMD = MTOT * DIM;
    const int nLM = VOC * DIM;

    // x0 = embed[tokens]; weights -> bf16
    k_embed<<<MTOT, 256, 0, stream>>>(tokens, embed, x);
    k_cvt_bf16<<<4096, 256, 0, stream>>>(Wk, Wk16, nDD);
    k_cvt_bf16<<<4096, 256, 0, stream>>>(Wv, Wv16, nDD);
    k_cvt_bf16<<<4096, 256, 0, stream>>>(Ws, Ws16, nDD);
    k_cvt_bf16<<<4096, 256, 0, stream>>>(Wo, Wo16, nDD);
    k_cvt_bf16<<<8192, 256, 0, stream>>>(lm_head, LM16, nLM);

    dim3 gD(DIM / 128, MTOT / 64);          // (8, 32)
    dim3 gAttn(SEQ / 16, NH, NB);           // (64, 8, 2)

    for (int it = 0; it < NIT; it++) {
        k_rmsnorm<<<MTOT, 256, 0, stream>>>(x, norm_w + it * DIM,
                                            gamma + it * DIM, beta + it * DIM, hb);
        k_gemm<<<gD, 128, 0, stream>>>(hb, Wk16, DIM, DIM, Kb, nullptr, nullptr, nullptr);
        k_gemm<<<gD, 128, 0, stream>>>(hb, Wv16, DIM, DIM, Vb, nullptr, nullptr, nullptr);
        k_gemm<<<gD, 128, 0, stream>>>(hb, Ws16, DIM, DIM, nullptr, SWb, nullptr, nullptr);
        k_transposeV<<<nMD / 256, 256, 0, stream>>>(Vb, Vt);
        k_attn<<<gAttn, 32, 0, stream>>>(Kb, Vt, Pb);
        k_add_bf16<<<nMD / 256, 256, 0, stream>>>(Pb, SWb, Tb, nMD);
        k_gemm<<<gD, 128, 0, stream>>>(Tb, Wo16, DIM, DIM, nullptr, nullptr, x, step_size + it);
    }

    // final norm + lm_head
    k_rmsnorm<<<MTOT, 256, 0, stream>>>(x, out_norm, nullptr, nullptr, xb);
    dim3 gLM((VOC + 127) / 128, MTOT / 64); // (393, 32)
    k_gemm<<<gLM, 128, 0, stream>>>(xb, LM16, VOC, DIM, nullptr, out, nullptr, nullptr);
    (void)in_sizes; (void)n_in; (void)out_size; (void)ws_size;
}
